// PGAT_30562987278722
// MI455X (gfx1250) — compile-verified, matrix operand order checked by reading, not run on hardware
//
#include <hip/hip_runtime.h>
#include <hip/hip_bf16.h>

// ---------------------------------------------------------------------------
// Problem constants (B, M, N, E, H, D) = (8, 128, 512, 256, 8, 32)
// ---------------------------------------------------------------------------
constexpr int cB = 8;
constexpr int cM = 128;
constexpr int cN = 512;
constexpr int cE = 256;
constexpr int cH = 8;
constexpr int cD = 32;

typedef __attribute__((ext_vector_type(16))) __bf16        v16bf;
typedef __attribute__((ext_vector_type(8)))  float         v8f;
typedef __attribute__((ext_vector_type(2)))  float         v2f;
typedef __attribute__((ext_vector_type(4)))  unsigned int  u32x4;
typedef __attribute__((ext_vector_type(4)))  float         f32x4;

union BFrag { u32x4 u[2]; v16bf v; };

__device__ __forceinline__ unsigned short f2bf(float f) {
    unsigned int u = __float_as_uint(f);
    unsigned int r = (u + 0x7fffu + ((u >> 16) & 1u)) >> 16;
    return (unsigned short)r;
}

__device__ __forceinline__ v8f vzero8() {
    v8f z = {0.f, 0.f, 0.f, 0.f, 0.f, 0.f, 0.f, 0.f};
    return z;
}

// ---------------------------------------------------------------------------
// Kernel 1: per-row LayerNorm (E=256, one row per 256-thread block) -> bf16
// ---------------------------------------------------------------------------
__global__ __launch_bounds__(256)
void ln_bf16_kernel(const float* __restrict__ x,
                    const float* __restrict__ sc,
                    const float* __restrict__ bi,
                    unsigned short* __restrict__ y)
{
    const int row = blockIdx.x;
    const int t   = threadIdx.x;
    const float v = x[(size_t)row * cE + t];

    __shared__ float s1[256];
    __shared__ float s2[256];
    s1[t] = v; s2[t] = v * v;
    __syncthreads();
    for (int o = 128; o > 0; o >>= 1) {
        if (t < o) { s1[t] += s1[t + o]; s2[t] += s2[t + o]; }
        __syncthreads();
    }
    const float mean = s1[0] * (1.f / cE);
    const float var  = s2[0] * (1.f / cE) - mean * mean;
    const float rs   = rsqrtf(var + 1e-6f);
    y[(size_t)row * cE + t] = f2bf((v - mean) * rs * sc[t] + bi[t]);
}

// ---------------------------------------------------------------------------
// Kernel 2: transpose-convert W (E x E, row-major f32) -> Wt[n*E + k] bf16
// so that the WMMA B operand loads K-contiguous per lane.
// ---------------------------------------------------------------------------
__global__ __launch_bounds__(256)
void wtrans_kernel(const float* __restrict__ W, unsigned short* __restrict__ Wt)
{
    const int i = blockIdx.x * 256 + threadIdx.x;   // i = k*E + n
    const int k = i >> 8;
    const int n = i & 255;
    Wt[n * cE + k] = f2bf(W[i]);
}

// ---------------------------------------------------------------------------
// Kernel 3: bf16 WMMA GEMM  C[rows x 256] = A[rows x 256] @ W[256 x 256] + bias
//   A: bf16 row-major (ld=256).  Bt: bf16, Bt[n*256 + k] = W[k][n].
//   Block = 8 waves (2x4): 32 rows x 256 cols per block. Wave: 16 x 64 tile.
//   K loop: 8 steps of 32 via v_wmma_f32_16x16x32_bf16.
// Per ISA 16-bit A layout: lane l holds row m=l&15,
//   K elements {base..base+7, base+16..base+23}, base = (l>>4)*8
//   -> two contiguous 16B loads. B operand mirrored over columns.
// ---------------------------------------------------------------------------
__global__ __launch_bounds__(256)
void gemm_bf16_kernel(const unsigned short* __restrict__ A,
                      const unsigned short* __restrict__ Bt,
                      const float* __restrict__ bias,
                      float* __restrict__ C)
{
    const int tid = threadIdx.x;
    const int w   = tid >> 5;
    const int l   = tid & 31;
    const int wr  = w >> 2;        // 0..1
    const int wc  = w & 3;         // 0..3
    const int m   = l & 15;
    const int kb8 = (l >> 4) * 8;  // K sub-base within a 32-chunk
    const int row0 = blockIdx.x * 32 + wr * 16;

    const unsigned short* arow = A + (size_t)(row0 + m) * cE;

    v8f acc[4];
    acc[0] = vzero8(); acc[1] = vzero8(); acc[2] = vzero8(); acc[3] = vzero8();

    for (int kb = 0; kb < 8; ++kb) {
        const int k0 = kb * 32 + kb8;
        BFrag af;
        af.u[0] = *(const u32x4*)(arow + k0);
        af.u[1] = *(const u32x4*)(arow + k0 + 16);
        #pragma unroll
        for (int c = 0; c < 4; ++c) {
            const int n = wc * 64 + c * 16 + m;
            const unsigned short* brow = Bt + (size_t)n * cE;
            BFrag bf;
            bf.u[0] = *(const u32x4*)(brow + k0);
            bf.u[1] = *(const u32x4*)(brow + k0 + 16);
            acc[c] = __builtin_amdgcn_wmma_f32_16x16x32_bf16(
                false, af.v, false, bf.v, (short)0, acc[c], false, false);
        }
    }

    // C/D layout: VGPR r, lane l -> row = r + (l>>4)*8, col = l&15
    const int lr = (l >> 4) * 8;
    #pragma unroll
    for (int c = 0; c < 4; ++c) {
        const int n = wc * 64 + c * 16 + m;
        const float bn = bias[n];
        #pragma unroll
        for (int r = 0; r < 8; ++r) {
            C[(size_t)(row0 + r + lr) * cE + n] = acc[c][r] + bn;
        }
    }
}

// ---------------------------------------------------------------------------
// Kernel 4: fused additive attention.
//   Grid: B * (M/16) = 64 blocks. 8 waves/block; wave h owns head h.
//   Flash-style online softmax over N in chunks of 32.
//   Output accumulation P(16 x nc) @ K(nc x 32) via v_wmma_f32_16x16x4_f32.
// Lane mapping (matches f32 16x16x4 A/B layouts):
//   logits: lane l computes row m=l&15, columns j with (j&3) in {2*(l>>4), 2*(l>>4)+1}
//   A step k: lane supplies P[m, 4k + 2*(l>>4) + {0,1}]
//   B step k: lane supplies K[4k + 2*(l>>4) + {0,1}, d=l&15 (+16 for hi tile)]
// ---------------------------------------------------------------------------
__global__ __launch_bounds__(256)
void attn_kernel(const float* __restrict__ qproj,
                 const float* __restrict__ kproj,
                 const float* __restrict__ skip,
                 const unsigned char* __restrict__ qmask,
                 const unsigned char* __restrict__ kmask,
                 const float* __restrict__ avec,
                 float* __restrict__ out)
{
    constexpr float FMIN = -3.402823466e38f;
    const int b    = blockIdx.x >> 3;
    const int m0   = (blockIdx.x & 7) * 16;
    const int tid  = threadIdx.x;
    const int h    = tid >> 5;
    const int l    = tid & 31;
    const int m    = l & 15;
    const int half = l >> 4;

    __shared__ float sq[cH][16][33];
    __shared__ float sk[cH][32][33];
    __shared__ float sa[cH][32];
    __shared__ float sscale[cH][16];
    __shared__ float srowsum[cH][16];

    // stage a[h, :] (thread tid belongs to wave h, writes its own head's slice)
    sa[h][l] = avec[h * cD + l];

    // stage q tile: wave h -> sq[h][m][d], lane covers d in [half*16, half*16+16)
    {
        const float* qbase = qproj + (size_t)(b * cM + m0 + m) * cE + h * cD + half * 16;
        #pragma unroll
        for (int i = 0; i < 4; ++i) {
            f32x4 v = *(const f32x4*)(qbase + i * 4);
            sq[h][m][half * 16 + i * 4 + 0] = v[0];
            sq[h][m][half * 16 + i * 4 + 1] = v[1];
            sq[h][m][half * 16 + i * 4 + 2] = v[2];
            sq[h][m][half * 16 + i * 4 + 3] = v[3];
        }
    }
    __syncthreads();

    const bool qv = qmask[b * cM + m0 + m] != 0;

    v8f acc0 = vzero8();
    v8f acc1 = vzero8();
    float row_max = -INFINITY;
    float row_sum = 0.f;

    for (int nc = 0; nc < 16; ++nc) {
        const int n0 = nc * 32;

        // stage K chunk: lane loads row j = l, all 32 d for this head
        {
            const float* krow = kproj + (size_t)(b * cN + n0 + l) * cE + h * cD;
            #pragma unroll
            for (int i = 0; i < 8; ++i) {
                f32x4 v = *(const f32x4*)(krow + i * 4);
                sk[h][l][i * 4 + 0] = v[0];
                sk[h][l][i * 4 + 1] = v[1];
                sk[h][l][i * 4 + 2] = v[2];
                sk[h][l][i * 4 + 3] = v[3];
            }
            if (nc < 15) {
                __builtin_prefetch(kproj + (size_t)(b * cN + n0 + 32 + l) * cE + h * cD, 0, 3);
            }
        }

        // logits: ps[i] = sum_d a[h,d] * mish(q[m,d] + k[j,d]),  j per lane slot
        float ps[16];
        #pragma unroll
        for (int i = 0; i < 16; ++i) {
            const int j = (i >> 1) * 4 + half * 2 + (i & 1);
            const bool kv = kmask[b * cN + n0 + j] != 0;
            float s = 0.f;
            for (int d = 0; d < cD; ++d) {
                const float xv = sq[h][m][d] + sk[h][j][d];
                const float e  = __expf(fminf(xv, 15.f));
                const float u  = e * (e + 2.f);
                const float mi = xv * u * __builtin_amdgcn_rcpf(u + 2.f);
                s = fmaf(sa[h][d], mi, s);
            }
            ps[i] = (qv && kv) ? s : FMIN;
        }

        // online softmax stats (row m lives in lanes m and m+16)
        float mx = ps[0];
        #pragma unroll
        for (int i = 1; i < 16; ++i) mx = fmaxf(mx, ps[i]);
        mx = fmaxf(mx, __shfl_xor(mx, 16));
        const float nm  = fmaxf(row_max, mx);
        const float scl = __expf(row_max - nm);
        row_max = nm;
        float ls = 0.f;
        #pragma unroll
        for (int i = 0; i < 16; ++i) { ps[i] = __expf(ps[i] - nm); ls += ps[i]; }
        ls += __shfl_xor(ls, 16);
        row_sum = row_sum * scl + ls;

        // broadcast per-row rescale to the accumulator row layout via wave LDS
        if (l < 16) sscale[h][m] = scl;
        #pragma unroll
        for (int r = 0; r < 8; ++r) {
            const float s2 = sscale[h][r + half * 8];
            acc0[r] *= s2;
            acc1[r] *= s2;
        }

        // accumulate O += P @ K with f32 WMMA 16x16x4 (8 K-steps of 4)
        const int kk = half * 2;
        #pragma unroll
        for (int k = 0; k < 8; ++k) {
            v2f A; A[0] = ps[2 * k]; A[1] = ps[2 * k + 1];
            const int j0 = 4 * k + kk;
            v2f B0; B0[0] = sk[h][j0][m];      B0[1] = sk[h][j0 + 1][m];
            v2f B1; B1[0] = sk[h][j0][m + 16]; B1[1] = sk[h][j0 + 1][m + 16];
            acc0 = __builtin_amdgcn_wmma_f32_16x16x4_f32(
                false, A, false, B0, (short)0, acc0, false, false);
            acc1 = __builtin_amdgcn_wmma_f32_16x16x4_f32(
                false, A, false, B1, (short)0, acc1, false, false);
        }
    }

    // epilogue: normalize, add skip, store
    if (l < 16) srowsum[h][m] = row_sum;
    #pragma unroll
    for (int r = 0; r < 8; ++r) {
        const int mr = r + half * 8;
        const float inv = __builtin_amdgcn_rcpf(srowsum[h][mr]);
        const size_t base = (size_t)(b * cM + m0 + mr) * cE + h * cD;
        out[base + m]      = acc0[r] * inv + skip[base + m];
        out[base + m + 16] = acc1[r] * inv + skip[base + m + 16];
    }
}

// ---------------------------------------------------------------------------
// Launch
// ---------------------------------------------------------------------------
extern "C" void kernel_launch(void* const* d_in, const int* in_sizes, int n_in,
                              void* d_out, int out_size, void* d_ws, size_t ws_size,
                              hipStream_t stream)
{
    (void)in_sizes; (void)n_in; (void)out_size; (void)ws_size;

    const float*         query      = (const float*)d_in[0];
    const float*         key        = (const float*)d_in[1];
    const unsigned char* qmask      = (const unsigned char*)d_in[2];  // bool8
    const unsigned char* kmask      = (const unsigned char*)d_in[3];  // bool8
    const float*         ln_q_scale = (const float*)d_in[4];
    const float*         ln_q_bias  = (const float*)d_in[5];
    const float*         ln_k_scale = (const float*)d_in[6];
    const float*         ln_k_bias  = (const float*)d_in[7];
    const float*         Wq         = (const float*)d_in[8];
    const float*         bq         = (const float*)d_in[9];
    const float*         Wk         = (const float*)d_in[10];
    const float*         bk         = (const float*)d_in[11];
    const float*         avec       = (const float*)d_in[12];
    float*               out        = (float*)d_out;

    // workspace layout (bytes)
    char* w = (char*)d_ws;
    unsigned short* qn    = (unsigned short*)(w + 0);        //  512 KB: LN(query) bf16
    unsigned short* kn    = (unsigned short*)(w + 524288);   // 2048 KB: LN(key) bf16
    unsigned short* wqt   = (unsigned short*)(w + 2621440);  //  128 KB: Wq^T bf16
    unsigned short* wkt   = (unsigned short*)(w + 2752512);  //  128 KB: Wk^T bf16
    float*          qproj = (float*)(w + 2883584);           // 1024 KB
    float*          kproj = (float*)(w + 3932160);           // 4096 KB

    ln_bf16_kernel<<<cB * cM, 256, 0, stream>>>(query, ln_q_scale, ln_q_bias, qn);
    ln_bf16_kernel<<<cB * cN, 256, 0, stream>>>(key,   ln_k_scale, ln_k_bias, kn);
    wtrans_kernel<<<(cE * cE) / 256, 256, 0, stream>>>(Wq, wqt);
    wtrans_kernel<<<(cE * cE) / 256, 256, 0, stream>>>(Wk, wkt);
    gemm_bf16_kernel<<<(cB * cM) / 32, 256, 0, stream>>>(qn, wqt, bq, qproj);
    gemm_bf16_kernel<<<(cB * cN) / 32, 256, 0, stream>>>(kn, wkt, bk, kproj);
    attn_kernel<<<cB * (cM / 16), 256, 0, stream>>>(qproj, kproj, query,
                                                    qmask, kmask, avec, out);
}
